// SpikingS4Layer_25254407700685
// MI455X (gfx1250) — matrix operand order repaired and checked by reading, not compiled, not run on hardware
//
#include <hip/hip_runtime.h>
#include <hip/hip_bf16.h>

typedef _Float16 v16h __attribute__((ext_vector_type(16)));
typedef _Float16 v8h  __attribute__((ext_vector_type(8)));
typedef float    v8f  __attribute__((ext_vector_type(8)));

#define T_LEN 8192
#define B_SZ  16
#define F_SZ  32
#define L_SZ  16
#define S_TAPS 10
#define OUT_F 128
#define TP (T_LEN + S_TAPS - 1)   // padded time length (9 zero rows in front per b)
#define KTOT (S_TAPS * F_SZ)      // 320

// ---------------------------------------------------------------------------
// Kernel 1: fold em_w/dec_w/dec_b/em_b into the conv filter, emitted directly
// in WMMA B-fragment order (fp16), plus the fused fp32 bias.
// VhB flat index: ((s*8 + tile)*32 + lane)*16 + j
//   lane = hi*16 + n ; element j <-> K = s*32 + hi*16 + j ; N = tile*16 + n
// ---------------------------------------------------------------------------
__global__ __launch_bounds__(256) void fold_kernel(
    const float* __restrict__ K, const float* __restrict__ em_w,
    const float* __restrict__ em_b, const float* __restrict__ dec_w,
    const float* __restrict__ dec_b, _Float16* __restrict__ VhB,
    float* __restrict__ bias) {
  __shared__ float W[L_SZ * KTOT];  // W[l][s][f] = sum_o em_w[l,o]*K[l,s,o,f]
  const int tid = threadIdx.x;

  for (int idx = tid; idx < L_SZ * KTOT; idx += 256) {
    int l = idx / KTOT;
    int rem = idx - l * KTOT;   // s*32 + f
    int s = rem >> 5, f = rem & 31;
    const float* Kp = K + (((size_t)(l * S_TAPS + s)) * F_SZ) * F_SZ + f;  // K[l][s][o][f]
    const float* ew = em_w + l * F_SZ;
    float acc = 0.f;
#pragma unroll
    for (int o = 0; o < F_SZ; ++o) acc += ew[o] * Kp[(size_t)o * F_SZ];
    W[idx] = acc;
  }
  __syncthreads();

  if (tid < OUT_F) {
    float acc = dec_b[tid];
#pragma unroll
    for (int l = 0; l < L_SZ; ++l) acc += dec_w[tid * L_SZ + l] * em_b[l];
    bias[tid] = acc;
  }

  for (int e = tid; e < S_TAPS * 8 * 32 * 16; e += 256) {   // 40960 entries
    int j    = e & 15;
    int lane = (e >> 4) & 31;
    int tile = (e >> 9) & 7;
    int s    = e >> 12;
    int hi = lane >> 4, n = lane & 15;
    int f = hi * 16 + j;          // feature index within tap s
    int u = tile * 16 + n;        // output feature
    float acc = 0.f;
#pragma unroll
    for (int l = 0; l < L_SZ; ++l)
      acc += dec_w[u * L_SZ + l] * W[l * KTOT + s * F_SZ + f];
    VhB[e] = (_Float16)acc;
  }
}

// ---------------------------------------------------------------------------
// Kernel 2: fp32 -> fp16 conversion of x with 9 zero rows of causal padding
// per batch. xh layout: [b][tp][f], tp = t + 9.
// ---------------------------------------------------------------------------
__global__ __launch_bounds__(256) void cvt_kernel(const float* __restrict__ x,
                                                  _Float16* __restrict__ xh) {
  const size_t total = (size_t)B_SZ * TP * F_SZ;
  for (size_t idx = (size_t)blockIdx.x * blockDim.x + threadIdx.x; idx < total;
       idx += (size_t)gridDim.x * blockDim.x) {
    size_t f  = idx & 31;
    size_t tp = (idx >> 5) % TP;
    size_t b  = idx / ((size_t)TP * F_SZ);
    long   t  = (long)tp - (S_TAPS - 1);
    float  v  = (t >= 0) ? x[((size_t)t * B_SZ + b) * F_SZ + f] : 0.f;
    xh[idx] = (_Float16)v;
  }
}

// ---------------------------------------------------------------------------
// Kernel 3: causal-conv GEMM via v_wmma_f32_16x16x32_f16.
// Wave -> one 16(t) x 16(u) output tile; block = 8 waves covering u=0..127.
// K = 320 = 10 taps x 32 features = 10 WMMA steps, fp32 accumulation.
// ---------------------------------------------------------------------------
__global__ __launch_bounds__(256) void conv_wmma_kernel(
    const _Float16* __restrict__ xh, const _Float16* __restrict__ VhB,
    const float* __restrict__ bias, float* __restrict__ out) {
  const int t0   = blockIdx.x * 16;
  const int b    = blockIdx.y;
  const int tile = threadIdx.x >> 5;   // which 16-wide u tile (0..7)
  const int lane = threadIdx.x & 31;
  const int m    = lane & 15;
  const int hi   = lane >> 4;

  // Preload all 10 B fragments (filter, identical for every block -> L2 hot).
  v16h bf[S_TAPS];
#pragma unroll
  for (int s = 0; s < S_TAPS; ++s)
    bf[s] = *(const v16h*)(VhB + (size_t)s * 4096 + ((size_t)tile * 32 + lane) * 16);

  // A row pointer for tap s=0: xh[b][t0 + m + 9][*]; tap s shifts back one row.
  const _Float16* px = xh + ((size_t)b * TP + (size_t)(t0 + m + S_TAPS - 1)) * F_SZ;

  v8f acc = {};
#pragma unroll
  for (int s = 0; s < S_TAPS; ++s) {
    v8h alo = *(const v8h*)(px + hi * 16);       // f = hi*16 .. hi*16+7
    v8h ahi = *(const v8h*)(px + hi * 16 + 8);   // f = hi*16+8 .. hi*16+15
    v16h a = __builtin_shufflevector(alo, ahi, 0, 1, 2, 3, 4, 5, 6, 7, 8, 9,
                                     10, 11, 12, 13, 14, 15);
    acc = __builtin_amdgcn_wmma_f32_16x16x32_f16(
        /*neg_a=*/false, a, /*neg_b=*/false, bf[s],
        /*c_mod=*/(short)0, acc, /*reuse_a=*/false, /*reuse_b=*/false);
    px -= F_SZ;  // next tap: one time-row earlier
  }

  // D layout: VGPR v -> (M = v + 8*hi, N = m). out[b][t][u], u contiguous.
  const int u = tile * 16 + m;
  const float bs = bias[u];
  size_t obase = (((size_t)b * T_LEN) + (size_t)(t0 + 8 * hi)) * OUT_F + u;
#pragma unroll
  for (int v = 0; v < 8; ++v)
    out[obase + (size_t)v * OUT_F] = acc[v] + bs;
}

// ---------------------------------------------------------------------------
extern "C" void kernel_launch(void* const* d_in, const int* in_sizes, int n_in,
                              void* d_out, int out_size, void* d_ws, size_t ws_size,
                              hipStream_t stream) {
  const float* x     = (const float*)d_in[0];
  const float* K     = (const float*)d_in[1];
  const float* em_w  = (const float*)d_in[2];
  const float* em_b  = (const float*)d_in[3];
  const float* dec_w = (const float*)d_in[4];
  const float* dec_b = (const float*)d_in[5];
  float* out = (float*)d_out;

  char* ws = (char*)d_ws;
  _Float16* VhB  = (_Float16*)ws;            // 40960 halves = 81920 B
  float*    bias = (float*)(ws + 81920);     // 128 floats  = 512 B
  _Float16* xh   = (_Float16*)(ws + 82432);  // B*TP*F halves = 8,397,824 B

  fold_kernel<<<1, 256, 0, stream>>>(K, em_w, em_b, dec_w, dec_b, VhB, bias);
  cvt_kernel<<<1024, 256, 0, stream>>>(x, xh);

  dim3 grid(T_LEN / 16, B_SZ);
  conv_wmma_kernel<<<grid, 256, 0, stream>>>(xh, VhB, bias, out);
}
